// Attention_5677946765904
// MI455X (gfx1250) — matrix operand order repaired and checked by reading, not compiled, not run on hardware
//
#include <hip/hip_runtime.h>
#include <hip/hip_bf16.h>

typedef __attribute__((ext_vector_type(16))) _Float16 v16h;
typedef __attribute__((ext_vector_type(8)))  float    v8f;
typedef unsigned int u32;
typedef __attribute__((ext_vector_type(4)))  u32 v4u;
typedef __attribute__((ext_vector_type(8)))  int v8i;
typedef __attribute__((ext_vector_type(4)))  int v4i;

#define WMMA_F16(A, B, C) \
    __builtin_amdgcn_wmma_f32_16x16x32_f16(false, (A), false, (B), (short)0, (C), false, false)

union AF {
    v16h v;
    unsigned int u[8];
    _Float16 h[16];
};

// K position this code packs into A VGPR j (contiguous order, B matches), given lane:
//   v0..v7: lanes0-15 K=0..15, lanes16-31 K=16..31
__device__ __forceinline__ int a_kpat(int j, int lane) {
    return (((lane >> 4) & 1) << 4) + (j << 1);
}

// ---------------- Problem constants (B=2, N=2048, C=1024, H=16, Dh=64) ---------
#define BATCH   2
#define SEQ     2048
#define DM      1024
#define NH      16
#define DH      64
#define ROWS    (BATCH * SEQ)        // 4096
#define QKV_N   (3 * DM)             // 3072

// ---------------- Tensor Data Mover: 2-D tile Global -> LDS --------------------
// Builds D# per CDNA5 ISA (group0 bits: count/lds_addr/global_addr/type=2;
// group1 bits: data_size=2B, tensor dims, tile dims, dim0 stride).  All units
// are elements (2-byte halves). 6-arg clang-23 builtin; groups 2/3 zero (2-D).
__device__ __forceinline__ void tdm_load_2d(u32 lds_byte_off, const _Float16* gptr,
                                            u32 tile_d0, u32 tile_d1,
                                            u32 tensor_d0, u32 tensor_d1,
                                            u32 stride0) {
    unsigned long long ga = (unsigned long long)(uintptr_t)gptr;
    v4u g0;
    g0[0] = 1u;                                            // count=1, user desc
    g0[1] = lds_byte_off;                                  // lds_addr (bytes)
    g0[2] = (u32)(ga & 0xffffffffu);                       // global_addr[31:0]
    g0[3] = (u32)((ga >> 32) & 0x01ffffffu) | (2u << 30);  // addr[56:32] | type=2
    v8i g1;
    g1[0] = (int)(1u << 16);                               // data_size=1 (2 bytes)
    g1[1] = (int)((tensor_d0 & 0xffffu) << 16);            // tensor_dim0[15:0]
    g1[2] = (int)(((tensor_d0 >> 16) & 0xffffu) |
                  ((tensor_d1 & 0xffffu) << 16));          // dim0 hi | dim1 lo
    g1[3] = (int)(((tensor_d1 >> 16) & 0xffffu) |
                  ((tile_d0 & 0xffffu) << 16));            // dim1 hi | tile_dim0
    g1[4] = (int)(tile_d1 & 0xffffu);                      // tile_dim1 (tile_dim2=0)
    g1[5] = (int)stride0;                                  // tensor_dim0_stride lo
    g1[6] = 0;                                             // stride0 hi | stride1 lo
    g1[7] = 0;
    v4i z4 = {0, 0, 0, 0};
    v8i z8 = {0, 0, 0, 0, 0, 0, 0, 0};
    __builtin_amdgcn_tensor_load_to_lds(g0, g1, z4, z4, z8, 0);
}

__device__ __forceinline__ u32 lds_offset_of(const void* p) {
    // generic-pointer low 32 bits == LDS byte offset (flat aperture truncation)
    return (u32)(uintptr_t)p;
}

// =============================== convert kernel ================================
__global__ void cvt_kernel(const float* __restrict__ x,
                           const float* __restrict__ wqkv,
                           const float* __restrict__ wproj,
                           _Float16* __restrict__ xh,
                           _Float16* __restrict__ wqkvT,
                           _Float16* __restrict__ wprojT) {
    const int XN = ROWS * DM;          // 4M
    const int WQ = QKV_N * DM;         // 3M
    const int WP = DM * DM;            // 1M
    const int total = XN + WQ + WP;
    for (int i = blockIdx.x * blockDim.x + threadIdx.x; i < total;
         i += gridDim.x * blockDim.x) {
        if (i < XN) {
            xh[i] = (_Float16)x[i];
        } else if (i < XN + WQ) {
            int t = i - XN;
            int n = t >> 10, k = t & 1023;          // wqkvT[n][k] = wqkv[k][n]
            wqkvT[t] = (_Float16)wqkv[(size_t)k * QKV_N + n];
        } else {
            int t = i - XN - WQ;
            int n = t >> 10, k = t & 1023;          // wprojT[n][k] = wproj[k][n]
            wprojT[t] = (_Float16)wproj[(size_t)k * DM + n];
        }
    }
}

// =============================== QKV GEMM ======================================
__global__ void qkv_gemm_kernel(const _Float16* __restrict__ xh,
                                const _Float16* __restrict__ wT,
                                const float* __restrict__ bias,
                                _Float16* __restrict__ qb,
                                _Float16* __restrict__ kb,
                                _Float16* __restrict__ vtb) {
    const int lane = threadIdx.x & 31;
    const int wid  = threadIdx.x >> 5;
    const int gid  = blockIdx.x * 8 + wid;           // 12288 waves
    const int NT   = QKV_N / 64;                     // 48
    const int mt   = gid / NT, nt = gid % NT;
    const int M0   = mt * 16, N0 = nt * 64;
    const int m    = lane & 15;
    const int khalf = ((lane >> 4) & 1) << 4;

    // single live base pointer per array; tile/pattern offsets are immediates
    const _Float16* aBase = xh + (size_t)(M0 + m) * DM;
    const _Float16* bBase = wT + (size_t)(N0 + m) * DM + khalf;

    auto loadA = [&](int k0, AF& a) {
        #pragma unroll
        for (int j = 0; j < 8; ++j)
            a.u[j] = *(const unsigned int*)(aBase + k0 + a_kpat(j, lane));
    };
    auto loadB = [&](int k0, AF* b) {
        #pragma unroll
        for (int t = 0; t < 4; ++t)
            #pragma unroll
            for (int j = 0; j < 8; ++j)
                b[t].u[j] = *(const unsigned int*)(bBase + (size_t)t * 16 * DM +
                                                   k0 + 2 * j);
    };

    v8f acc[4] = {};
    AF a0, a1, b0[4], b1[4];
    loadA(0, a0); loadB(0, b0);
    for (int k0 = 0; k0 < DM; k0 += 64) {            // ping-pong double buffer
        loadA(k0 + 32, a1); loadB(k0 + 32, b1);
        #pragma unroll
        for (int t = 0; t < 4; ++t) acc[t] = WMMA_F16(a0.v, b0[t].v, acc[t]);
        if (k0 + 64 < DM) { loadA(k0 + 64, a0); loadB(k0 + 64, b0); }
        #pragma unroll
        for (int t = 0; t < 4; ++t) acc[t] = WMMA_F16(a1.v, b1[t].v, acc[t]);
    }

    const int rbase = M0 + (((lane >> 4) & 1) << 3);
    #pragma unroll
    for (int t = 0; t < 4; ++t) {
        const int c = N0 + t * 16 + m;
        const float bs = bias[c];
        const int which = c >> 10;       // 0=Q 1=K 2=V
        const int cc = c & 1023;
        const int h = cc >> 6, d = cc & 63;
        #pragma unroll
        for (int j = 0; j < 8; ++j) {
            const int r  = rbase + j;
            const int bb = r >> 11, n = r & 2047;
            const float v = acc[t][j] + bs;
            const size_t hb = (size_t)(bb * NH + h);
            if (which == 0)
                qb[(hb * SEQ + n) * DH + d] = (_Float16)(v * 0.125f); // fold 1/sqrt(Dh)
            else if (which == 1)
                kb[(hb * SEQ + n) * DH + d] = (_Float16)v;
            else
                vtb[(hb * DH + d) * SEQ + n] = (_Float16)v;
        }
    }
}

// =============================== Flash attention ===============================
// 8 waves/WG cooperate on 8 query tiles of the same (b,h). K/V blocks stream
// into double-buffered LDS via the Tensor Data Mover, overlapped with WMMA.
__global__ void attn_kernel(const _Float16* __restrict__ qb,
                            const _Float16* __restrict__ kbuf,
                            const _Float16* __restrict__ vtb,
                            _Float16* __restrict__ omid) {
    __shared__ _Float16 Kb[2][32 * DH];      // [buf][key_local(32)][d(64)]  4KB each
    __shared__ _Float16 Vb[2][DH * 32];      // [buf][d(64)][key_local(32)]  4KB each
    __shared__ _Float16 pbuf[8][16 * 32];    // per-wave P staging            8KB

    const int lane = threadIdx.x & 31;
    const int wid  = threadIdx.x >> 5;
    const int QC   = SEQ / 128;                      // 16 query chunks per (b,h)
    const int bb   = blockIdx.x / (NH * QC);
    const int rem  = blockIdx.x % (NH * QC);
    const int h    = rem / QC;
    const int q0   = (rem % QC) * 128 + wid * 16;

    const int m     = lane & 15;
    const int hi    = (lane >> 4) & 1;
    const int khalf = hi << 4;

    const _Float16* qh  = qb   + ((size_t)(bb * NH + h) * SEQ) * DH;
    const _Float16* kh  = kbuf + ((size_t)(bb * NH + h) * SEQ) * DH;
    const _Float16* vth = vtb  + ((size_t)(bb * NH + h) * DH) * SEQ;

    // Q A-tiles (Dh chunks of 32)
    AF qa[2];
    {
        const _Float16* qp = qh + (size_t)(q0 + m) * DH;
        #pragma unroll
        for (int c = 0; c < 2; ++c)
            #pragma unroll
            for (int j = 0; j < 8; ++j)
                qa[c].u[j] = *(const unsigned int*)(qp + c * 32 + a_kpat(j, lane));
    }

    v8f acc[4] = {};
    float mstat[8], lstat[8];
    #pragma unroll
    for (int j = 0; j < 8; ++j) { mstat[j] = -3.0e38f; lstat[j] = 0.0f; }

    _Float16* pb = &pbuf[wid][0];

    // prologue: DMA key-block 0 into buffer 0
    if (wid == 0) {
        tdm_load_2d(lds_offset_of(&Kb[0][0]), kh,            /*tile*/ DH, 32,
                    /*tensor*/ DH, SEQ, /*stride0*/ DH);
        tdm_load_2d(lds_offset_of(&Vb[0][0]), vth,           /*tile*/ 32, DH,
                    /*tensor*/ SEQ, DH, /*stride0*/ SEQ);
        __builtin_amdgcn_s_wait_tensorcnt(0);
    }
    __syncthreads();

    const int NBLK = SEQ / 32;                       // 64
    for (int ib = 0; ib < NBLK; ++ib) {
        const int cur = ib & 1;
        // DMA next block into the other buffer while we compute
        if (wid == 0 && ib + 1 < NBLK) {
            const int kn = (ib + 1) * 32;
            tdm_load_2d(lds_offset_of(&Kb[cur ^ 1][0]), kh + (size_t)kn * DH,
                        DH, 32, DH, SEQ, DH);
            tdm_load_2d(lds_offset_of(&Vb[cur ^ 1][0]), vth + kn,
                        32, DH, SEQ, DH, SEQ);
        }

        // ---- S = Q @ K^T (two 16x16 C tiles) from LDS ----
        const _Float16* kL = &Kb[cur][0] + m * DH + khalf;
        v8f s[2] = {};
        #pragma unroll
        for (int half = 0; half < 2; ++half) {
            #pragma unroll
            for (int c = 0; c < 2; ++c) {
                AF bk;
                #pragma unroll
                for (int j = 0; j < 8; ++j)
                    bk.u[j] = *(const unsigned int*)(kL + half * 16 * DH + c * 32 +
                                                     2 * j);
                s[half] = WMMA_F16(qa[c].v, bk.v, s[half]);
            }
        }

        // ---- online softmax (rows live across 16-lane groups) ----
        float pr0[8], pr1[8];
        #pragma unroll
        for (int j = 0; j < 8; ++j) {
            float rmax = fmaxf(s[0][j], s[1][j]);
            #pragma unroll
            for (int off = 1; off < 16; off <<= 1)
                rmax = fmaxf(rmax, __shfl_xor(rmax, off, 32));
            const float nm = fmaxf(mstat[j], rmax);
            const float al = __expf(mstat[j] - nm);
            const float p0 = __expf(s[0][j] - nm);
            const float p1 = __expf(s[1][j] - nm);
            float rs = p0 + p1;
            #pragma unroll
            for (int off = 1; off < 16; off <<= 1)
                rs += __shfl_xor(rs, off, 32);
            lstat[j] = lstat[j] * al + rs;
            mstat[j] = nm;
            #pragma unroll
            for (int t = 0; t < 4; ++t) acc[t][j] *= al;
            pr0[j] = p0; pr1[j] = p1;
        }

        // ---- P: C-layout -> LDS -> A-layout (16x32 f16) ----
        const int rowo = hi << 3;
        #pragma unroll
        for (int j = 0; j < 8; ++j) {
            pb[(rowo + j) * 32 + m]      = (_Float16)pr0[j];
            pb[(rowo + j) * 32 + 16 + m] = (_Float16)pr1[j];
        }
        AF pa;
        #pragma unroll
        for (int j = 0; j < 8; ++j)
            pa.u[j] = *(const unsigned int*)(pb + m * 32 + a_kpat(j, lane));

        // ---- O += P @ V (4 d-tiles of 16) from LDS ----
        const _Float16* vL = &Vb[cur][0] + m * 32 + khalf;
        #pragma unroll
        for (int t = 0; t < 4; ++t) {
            AF bv;
            #pragma unroll
            for (int j = 0; j < 8; ++j)
                bv.u[j] = *(const unsigned int*)(vL + t * 16 * 32 + 2 * j);
            acc[t] = WMMA_F16(pa.v, bv.v, acc[t]);
        }

        // next block's DMA must be complete before anyone touches it
        if (wid == 0) __builtin_amdgcn_s_wait_tensorcnt(0);
        __syncthreads();
    }

    // ---- epilogue: O /= l, write as [B, N, C] f16 for the projection GEMM ----
    const int rowo = hi << 3;
    #pragma unroll
    for (int j = 0; j < 8; ++j) {
        const float inv = 1.0f / lstat[j];
        const int row = q0 + rowo + j;
        #pragma unroll
        for (int t = 0; t < 4; ++t) {
            const int cc = h * DH + t * 16 + m;
            omid[((size_t)(bb * SEQ + row)) * DM + cc] = (_Float16)(acc[t][j] * inv);
        }
    }
}

// =============================== Projection GEMM ===============================
__global__ void proj_gemm_kernel(const _Float16* __restrict__ oh,
                                 const _Float16* __restrict__ wT,
                                 const float* __restrict__ bias,
                                 float* __restrict__ out) {
    const int lane = threadIdx.x & 31;
    const int wid  = threadIdx.x >> 5;
    const int gid  = blockIdx.x * 8 + wid;           // 4096 waves
    const int NT   = DM / 64;                        // 16
    const int mt   = gid / NT, nt = gid % NT;
    const int M0   = mt * 16, N0 = nt * 64;
    const int m    = lane & 15;
    const int khalf = ((lane >> 4) & 1) << 4;

    const _Float16* aBase = oh + (size_t)(M0 + m) * DM;
    const _Float16* bBase = wT + (size_t)(N0 + m) * DM + khalf;

    auto loadA = [&](int k0, AF& a) {
        #pragma unroll
        for (int j = 0; j < 8; ++j)
            a.u[j] = *(const unsigned int*)(aBase + k0 + a_kpat(j, lane));
    };
    auto loadB = [&](int k0, AF* b) {
        #pragma unroll
        for (int t = 0; t < 4; ++t)
            #pragma unroll
            for (int j = 0; j < 8; ++j)
                b[t].u[j] = *(const unsigned int*)(bBase + (size_t)t * 16 * DM +
                                                   k0 + 2 * j);
    };

    v8f acc[4] = {};
    AF a0, a1, b0[4], b1[4];
    loadA(0, a0); loadB(0, b0);
    for (int k0 = 0; k0 < DM; k0 += 64) {
        loadA(k0 + 32, a1); loadB(k0 + 32, b1);
        #pragma unroll
        for (int t = 0; t < 4; ++t) acc[t] = WMMA_F16(a0.v, b0[t].v, acc[t]);
        if (k0 + 64 < DM) { loadA(k0 + 64, a0); loadB(k0 + 64, b0); }
        #pragma unroll
        for (int t = 0; t < 4; ++t) acc[t] = WMMA_F16(a1.v, b1[t].v, acc[t]);
    }

    const int rbase = M0 + (((lane >> 4) & 1) << 3);
    #pragma unroll
    for (int t = 0; t < 4; ++t) {
        const int c = N0 + t * 16 + m;
        const float bs = bias[c];
        #pragma unroll
        for (int j = 0; j < 8; ++j)
            out[(size_t)(rbase + j) * DM + c] = acc[t][j] + bs;
    }
}

// =============================== launch ========================================
extern "C" void kernel_launch(void* const* d_in, const int* in_sizes, int n_in,
                              void* d_out, int out_size, void* d_ws, size_t ws_size,
                              hipStream_t stream) {
    const float* x     = (const float*)d_in[0];
    const float* wqkv  = (const float*)d_in[1];
    const float* bqkv  = (const float*)d_in[2];
    const float* wproj = (const float*)d_in[3];
    const float* bproj = (const float*)d_in[4];
    float* out = (float*)d_out;

    char* w = (char*)d_ws;
    _Float16* xh     = (_Float16*)w; w += (size_t)ROWS * DM * 2;       //  8 MB
    _Float16* wqkvT  = (_Float16*)w; w += (size_t)QKV_N * DM * 2;      //  6 MB
    _Float16* wprojT = (_Float16*)w; w += (size_t)DM * DM * 2;         //  2 MB
    _Float16* qb     = (_Float16*)w; w += (size_t)ROWS * DM * 2;       //  8 MB
    _Float16* kb     = (_Float16*)w; w += (size_t)ROWS * DM * 2;       //  8 MB
    _Float16* vtb    = (_Float16*)w; w += (size_t)ROWS * DM * 2;       //  8 MB
    _Float16* omid   = (_Float16*)w;                                   //  8 MB

    cvt_kernel<<<2048, 256, 0, stream>>>(x, wqkv, wproj, xh, wqkvT, wprojT);

    // 4096/16 * 3072/64 = 12288 waves -> 1536 blocks of 8 waves
    qkv_gemm_kernel<<<1536, 256, 0, stream>>>(xh, wqkvT, bqkv, qb, kb, vtb);

    // 2*16*(2048/128) = 512 blocks, 8 cooperating waves each
    attn_kernel<<<512, 256, 0, stream>>>(qb, kb, vtb, omid);

    // 4096/16 * 1024/64 = 4096 waves -> 512 blocks
    proj_gemm_kernel<<<512, 256, 0, stream>>>(omid, wprojT, bproj, out);
}
